// RBP_27058293964894
// MI455X (gfx1250) — compile-verified, matrix operand order checked by reading, not run on hardware
//
#include <hip/hip_runtime.h>
#include <hip/hip_bf16.h>
#include <math.h>

// ---------------------------------------------------------------------------
// CDNA5 (gfx1250) wave32 WMMA types
// ---------------------------------------------------------------------------
typedef __attribute__((ext_vector_type(16))) __bf16 v16bf;
typedef __attribute__((ext_vector_type(8)))  float  v8f;

#define P_PAIRS 256
#define B_IMG   16
#define L_MAX   24
#define DREL    1936

// Convert 16 floats (optional relu) to a bf16 WMMA operand.
__device__ __forceinline__ v16bf to_bf16x16(const float* f, bool doRelu) {
  v16bf v;
#pragma unroll
  for (int j = 0; j < 16; ++j) {
    float x = f[j];
    if (doRelu) x = fmaxf(x, 0.0f);
    v[j] = (__bf16)x;
  }
  return v;
}

// Full 32-K step, unit stride: two b128 loads per 8-float run.
// off = rbase + k0, 16-byte aligned by construction.
__device__ __forceinline__ v16bf load_full_vec(const float* __restrict__ base, long off,
                                               bool doRelu) {
  const float4* p = (const float4*)(base + off);
  float4 x0 = p[0];   // k0+0..3
  float4 x1 = p[1];   // k0+4..7
  float4 y0 = p[4];   // k0+16..19
  float4 y1 = p[5];   // k0+20..23
  float f[16] = {x0.x, x0.y, x0.z, x0.w, x1.x, x1.y, x1.z, x1.w,
                 y0.x, y0.y, y0.z, y0.w, y1.x, y1.y, y1.z, y1.w};
  return to_bf16x16(f, doRelu);
}

// Full 32-K step, fixed stride 49 (NCHW 1x1-conv input), 32-bit address math.
__device__ __forceinline__ v16bf load_full_s49(const float* __restrict__ base, int rbase,
                                               int k0) {
  float f[16];
  int o0 = rbase + k0 * 49;
#pragma unroll
  for (int j = 0; j < 8; ++j) {
    f[j]     = base[o0 + j * 49];
    f[j + 8] = base[o0 + (16 + j) * 49];
  }
  return to_bf16x16(f, false);
}

// K-tail step: clamp k into range (address always valid), mask invalid k by *0.
__device__ __forceinline__ v16bf load_tail(const float* __restrict__ base, long rbase,
                                           long kstride, int k0, int K, bool doRelu) {
  float f[16];
#pragma unroll
  for (int j = 0; j < 8; ++j) {
    int ka = k0 + j;
    int kb = k0 + 16 + j;
    int kca = ka < K - 1 ? ka : K - 1;
    int kcb = kb < K - 1 ? kb : K - 1;
    f[j]     = base[rbase + (long)kca * kstride] * ((ka < K) ? 1.0f : 0.0f);
    f[j + 8] = base[rbase + (long)kcb * kstride] * ((kb < K) ? 1.0f : 0.0f);
  }
  return to_bf16x16(f, doRelu);
}

// ---------------------------------------------------------------------------
// Generic bf16-WMMA GEMM:  C[m,n] = act(sum_k A[m,k] * B[n,k] + bias[n])
//   aMode: 0 = A[m*lda + k]
//          1 = A[idx[m*aIdxStride+aIdxOff]*lda + k]          (row gather)
//          2 = A[(m/49)*(lda*49) + k*49 + (m%49)]            (NCHW 1x1-conv input)
//   cMode: 0 = C[m*ldc + cColOff + n]
//          1 = C[(m/49)*12544 + n*49 + (m%49)]               (NCHW output)
// One wave per 16x16 tile, 8 waves per block, branchless inner loop.
// Out-of-range rows/cols use clamped (in-bounds) addresses; their garbage
// only lands in output rows/cols that the store guards discard.
// ---------------------------------------------------------------------------
__global__ void gemm_bf16_wmma(const float* __restrict__ A, long lda, int aMode,
                               const int* __restrict__ aIdx, int aIdxStride, int aIdxOff,
                               int aRelu,
                               const float* __restrict__ B, long ldb,
                               const float* __restrict__ bias,
                               float* __restrict__ C, long ldc, int cColOff, int cMode,
                               int cSig,
                               int M, int N, int K, int tilesM, int tilesN) {
  int wave = threadIdx.x >> 5;
  int lane = threadIdx.x & 31;
  int kh   = lane >> 4;   // K-half selector
  int mloc = lane & 15;

  int tile = blockIdx.x * (blockDim.x >> 5) + wave;
  if (tile >= tilesM * tilesN) return;   // wave-uniform exit keeps EXEC full for WMMA

  int tm = tile / tilesN;
  int tn = tile % tilesN;
  int mBase = tm * 16;
  int nBase = tn * 16;

  int  mRow  = mBase + mloc;
  int  mRowC = (mRow < M - 1) ? mRow : (M - 1);   // clamped: address always valid
  bool doRel = (aRelu != 0);

  long arb;
  long akstride = 1;
  if (aMode == 0) {
    arb = (long)mRowC * lda;
  } else if (aMode == 1) {
    int r = aIdx[mRowC * aIdxStride + aIdxOff];
    arb = (long)r * lda;
  } else {
    int n49 = mRowC / 49;
    int hw  = mRowC - n49 * 49;
    arb = (long)n49 * (lda * 49) + hw;
    akstride = 49;
  }

  int  nCol  = nBase + mloc;
  int  nColC = (nCol < N - 1) ? nCol : (N - 1);
  long brb   = (long)nColC * ldb;

  int Kmain = K & ~31;
  v8f acc = {};
  if (aMode != 2) {
    for (int kb = 0; kb < Kmain; kb += 32) {
      int k0 = kb + kh * 8;
      v16bf a = load_full_vec(A, arb + k0, doRel);
      v16bf b = load_full_vec(B, brb + k0, false);
      acc = __builtin_amdgcn_wmma_f32_16x16x32_bf16(false, a, false, b,
                                                    (short)0, acc, false, false);
    }
  } else {
    int arbi = (int)arb;     // union_feat offsets fit in 32 bits
    for (int kb = 0; kb < Kmain; kb += 32) {
      int k0 = kb + kh * 8;
      v16bf a = load_full_s49(A, arbi, k0);
      v16bf b = load_full_vec(B, brb + k0, false);
      acc = __builtin_amdgcn_wmma_f32_16x16x32_bf16(false, a, false, b,
                                                    (short)0, acc, false, false);
    }
  }
  if (Kmain < K) {
    int k0 = Kmain + kh * 8;
    v16bf a = load_tail(A, arb, akstride, k0, K, doRel);
    v16bf b = load_tail(B, brb, 1,        k0, K, false);
    acc = __builtin_amdgcn_wmma_f32_16x16x32_bf16(false, a, false, b,
                                                  (short)0, acc, false, false);
  }

  // Epilogue. C/D layout: lane l holds col nBase+(l&15); VGPR j -> row mBase+j+8*(l>>4)
  if (nCol < N) {
    float bs = bias ? bias[nCol] : 0.0f;
#pragma unroll
    for (int j = 0; j < 8; ++j) {
      int row = mBase + j + 8 * kh;
      if (row < M) {
        float v = acc[j] + bs;
        if (cSig) v = 1.0f / (1.0f + expf(-v));
        long addr;
        if (cMode == 0) {
          addr = (long)row * ldc + cColOff + nCol;
        } else {
          int n49 = row / 49;
          int hw  = row - n49 * 49;
          addr = (long)n49 * 12544 + (long)nCol * 49 + hw;
        }
        C[addr] = v;
      }
    }
  }
}

// ---------------------------------------------------------------------------
// Bilinear: bil[n,o] = sum_h sum_t s[n,h] * W[o,h,t] * o[n,t] + bil_b[o]
// Per block (nblk in [0,16), o in [0,712)):
//   phase 1 (WMMA): tmp[n,h] = sum_t o_mat[n,t] * W[o,h,t]    (16 x 712, K = 712)
//   phase 2 (VALU): bil[n]   = sum_h s[n,h] * tmp[n,h]
// so_cat row layout: [ s(0:712) | o(712:1424) ], ld = 1424.
// Writes into zb[:, 712 + o] so relu(concat(z,bil)) is contiguous for proj.
// Clamped h-columns >711 write garbage only into LDS padding cols 712..719,
// which phase 2 never reads.
// ---------------------------------------------------------------------------
__global__ void bilinear_wmma(const float* __restrict__ so_cat,
                              const float* __restrict__ bil_w,
                              const float* __restrict__ bil_b,
                              float* __restrict__ zb) {
  __shared__ float tmp[16][736];

  int nblk = blockIdx.x;   // 0..15
  int o    = blockIdx.y;   // 0..711
  int wave = threadIdx.x >> 5;
  int lane = threadIdx.x & 31;
  int kh   = lane >> 4;
  int mloc = lane & 15;

  const float* W    = bil_w + (size_t)o * 712 * 712;
  const float* Amat = so_cat + 712;                       // o-part of so_cat
  long arb = (long)(nblk * 16 + mloc) * 1424;             // A row base (always valid)

  for (int ht = wave; ht < 45; ht += 8) {                 // 712 -> 45 h-tiles
    int  hBase = ht * 16;
    int  hCol  = hBase + mloc;
    int  hColC = (hCol < 711) ? hCol : 711;
    long brb   = (long)hColC * 712;

    v8f acc = {};
    for (int kb = 0; kb < 704; kb += 32) {                // 22 full K-steps
      int k0 = kb + kh * 8;
      if (kb + 256 < 712)                                 // stream-ahead on the 1.44GB W
        __builtin_prefetch(W + brb + kb + 256, 0, 1);
      v16bf a = load_full_vec(Amat, arb + k0, false);
      v16bf b = load_full_vec(W,    brb + k0, false);
      acc = __builtin_amdgcn_wmma_f32_16x16x32_bf16(false, a, false, b,
                                                    (short)0, acc, false, false);
    }
    {                                                     // K tail: 704..711
      int k0 = 704 + kh * 8;
      v16bf a = load_tail(Amat, arb, 1, k0, 712, false);
      v16bf b = load_tail(W,    brb, 1, k0, 712, false);
      acc = __builtin_amdgcn_wmma_f32_16x16x32_bf16(false, a, false, b,
                                                    (short)0, acc, false, false);
    }
    int c = hBase + mloc;                                 // < 720, fits padded LDS
#pragma unroll
    for (int j = 0; j < 8; ++j) tmp[j + 8 * kh][c] = acc[j];
  }
  __syncthreads();

  // phase 2: 16 threads per n-row, strided dot over h, shfl-reduce within 16 lanes
  int nl = threadIdx.x >> 4;       // 0..15
  int c  = threadIdx.x & 15;
  int nglob = nblk * 16 + nl;
  const float* srow = so_cat + (long)nglob * 1424;
  float sum = 0.0f;
  for (int h = c; h < 712; h += 16) sum += tmp[nl][h] * srow[h];
  sum += __shfl_xor(sum, 1, 16);
  sum += __shfl_xor(sum, 2, 16);
  sum += __shfl_xor(sum, 4, 16);
  sum += __shfl_xor(sum, 8, 16);
  if (c == 0) zb[(long)nglob * 1424 + 712 + o] = sum + bil_b[o];
}

// ---------------------------------------------------------------------------
// conv1 (7x7 s2 p3) + relu + bn1 : [256,2,27,27] -> [256,128,14,14]
// ---------------------------------------------------------------------------
__global__ void conv1_relu_bn(const float* __restrict__ in, const float* __restrict__ w,
                              const float* __restrict__ bias,
                              const float* __restrict__ g, const float* __restrict__ bt,
                              const float* __restrict__ mn, const float* __restrict__ vr,
                              float* __restrict__ out) {
  int idx = blockIdx.x * blockDim.x + threadIdx.x;
  if (idx >= 256 * 128 * 14 * 14) return;
  int ox = idx % 14;
  int t  = idx / 14;
  int oy = t % 14;  t /= 14;
  int oc = t % 128;
  int n  = t / 128;
  float s = bias[oc];
  const float* wp = w  + oc * 2 * 49;
  const float* ip = in + (long)n * 2 * 729;
  for (int ic = 0; ic < 2; ++ic) {
    for (int ky = 0; ky < 7; ++ky) {
      int iy = oy * 2 - 3 + ky;
      if (iy < 0 || iy >= 27) continue;
      for (int kx = 0; kx < 7; ++kx) {
        int ix = ox * 2 - 3 + kx;
        if (ix < 0 || ix >= 27) continue;
        s += ip[ic * 729 + iy * 27 + ix] * wp[ic * 49 + ky * 7 + kx];
      }
    }
  }
  s = fmaxf(s, 0.0f);
  s = g[oc] * (s - mn[oc]) * rsqrtf(vr[oc] + 1e-5f) + bt[oc];
  out[idx] = s;
}

// maxpool 3x3 s2 p1 : [256,128,14,14] -> [256,128,7,7]
__global__ void maxpool_3s2(const float* __restrict__ in, float* __restrict__ out) {
  int idx = blockIdx.x * blockDim.x + threadIdx.x;
  if (idx >= 256 * 128 * 49) return;
  int ox = idx % 7;
  int t  = idx / 7;
  int oy = t % 7;   t /= 7;
  int oc = t % 128;
  int n  = t / 128;
  const float* ip = in + ((long)n * 128 + oc) * 196;
  float m = -INFINITY;
  for (int ky = 0; ky < 3; ++ky) {
    int iy = oy * 2 - 1 + ky;
    if (iy < 0 || iy >= 14) continue;
    for (int kx = 0; kx < 3; ++kx) {
      int ix = ox * 2 - 1 + kx;
      if (ix < 0 || ix >= 14) continue;
      m = fmaxf(m, ip[iy * 14 + ix]);
    }
  }
  out[idx] = m;
}

// conv2 (3x3 s1 p1) + relu + bn2 + add u : vrin = bn2(relu(conv2(x))) + u  [256,256,49]
__global__ void conv2_relu_bn_add(const float* __restrict__ in, const float* __restrict__ w,
                                  const float* __restrict__ bias,
                                  const float* __restrict__ g, const float* __restrict__ bt,
                                  const float* __restrict__ mn, const float* __restrict__ vr,
                                  const float* __restrict__ u, float* __restrict__ out) {
  int idx = blockIdx.x * blockDim.x + threadIdx.x;
  if (idx >= 256 * 256 * 49) return;
  int hw = idx % 49;
  int oc = (idx / 49) % 256;
  int n  = idx / (49 * 256);
  int oy = hw / 7, ox = hw % 7;
  float s = bias[oc];
  const float* wp = w  + (long)oc * 128 * 9;
  const float* ip = in + (long)n * 128 * 49;
  for (int ic = 0; ic < 128; ++ic) {
    const float* iq = ip + ic * 49;
    const float* wq = wp + ic * 9;
    for (int ky = 0; ky < 3; ++ky) {
      int iy = oy - 1 + ky;
      if (iy < 0 || iy >= 7) continue;
      for (int kx = 0; kx < 3; ++kx) {
        int ix = ox - 1 + kx;
        if (ix < 0 || ix >= 7) continue;
        s += iq[iy * 7 + ix] * wq[ky * 3 + kx];
      }
    }
  }
  s = fmaxf(s, 0.0f);
  s = g[oc] * (s - mn[oc]) * rsqrtf(vr[oc] + 1e-5f) + bt[oc];
  out[idx] = s + u[idx];
}

// so_cat[n] = [ subj(512) | emb1[labels[pi0]](200) | obj(512) | emb2[labels[pi1]](200) ]
__global__ void build_socat(const float* __restrict__ subj, const float* __restrict__ obj,
                            const float* __restrict__ emb1, const float* __restrict__ emb2,
                            const int* __restrict__ pair_idx, const int* __restrict__ labels,
                            float* __restrict__ so_cat) {
  int idx = blockIdx.x * blockDim.x + threadIdx.x;
  if (idx >= 256 * 1424) return;
  int n = idx / 1424;
  int c = idx % 1424;
  float v;
  if (c < 512)       v = subj[n * 512 + c];
  else if (c < 712)  v = emb1[labels[pair_idx[2 * n]] * 200 + (c - 512)];
  else if (c < 1224) v = obj[n * 512 + (c - 712)];
  else               v = emb2[labels[pair_idx[2 * n + 1]] * 200 + (c - 1224)];
  so_cat[idx] = v;
}

// counts[b], starts[b] from sorted im_idx (single tiny block)
__global__ void seg_counts(const int* __restrict__ im_idx, int* __restrict__ counts,
                           int* __restrict__ starts) {
  int b = threadIdx.x;
  if (b < B_IMG) {
    int c = 0, st = 0;
    for (int n = 0; n < P_PAIRS; ++n) {
      int v = im_idx[n];
      c  += (v == b);
      st += (v < b);
    }
    counts[b] = c;
    starts[b] = st;
  }
}

__global__ void seg_max_pool(const float* __restrict__ rel, const int* __restrict__ starts,
                             const int* __restrict__ counts, float* __restrict__ pooled) {
  int idx = blockIdx.x * blockDim.x + threadIdx.x;
  if (idx >= B_IMG * DREL) return;
  int b = idx / DREL;
  int d = idx % DREL;
  int s = starts[b];
  int e = s + counts[b];
  float m = -INFINITY;
  for (int n = s; n < e; ++n) m = fmaxf(m, rel[(long)n * DREL + d]);
  pooled[idx] = m;
}

__global__ void zero_f32(float* __restrict__ p, int n) {
  int i = blockIdx.x * blockDim.x + threadIdx.x;
  if (i < n) p[i] = 0.0f;
}

__global__ void scatter_memory(const float* __restrict__ rel, const int* __restrict__ im_idx,
                               const int* __restrict__ starts, float* __restrict__ mem) {
  int idx = blockIdx.x * blockDim.x + threadIdx.x;
  if (idx >= P_PAIRS * DREL) return;
  int n = idx / DREL;
  int d = idx % DREL;
  int b = im_idx[n];
  int pos = n - starts[b];
  mem[((long)pos * B_IMG + b) * DREL + d] = rel[idx];
}

__global__ void build_mask(const int* __restrict__ counts, float* __restrict__ mask) {
  int idx = blockIdx.x * blockDim.x + threadIdx.x;
  if (idx >= B_IMG * L_MAX) return;
  int b = idx / L_MAX;
  int l = idx % L_MAX;
  mask[idx] = (l >= counts[b]) ? 1.0f : 0.0f;
}

// ---------------------------------------------------------------------------
// Host-side launcher
// ---------------------------------------------------------------------------
static inline int cdiv(int a, int b) { return (a + b - 1) / b; }

static void run_gemm(hipStream_t s,
                     const float* A, long lda, int aMode, const int* aIdx, int aIdxStride,
                     int aIdxOff, int aRelu, const float* B, long ldb, const float* bias,
                     float* C, long ldc, int cColOff, int cMode, int cSig,
                     int M, int N, int K) {
  int tm = cdiv(M, 16), tn = cdiv(N, 16);
  int blocks = cdiv(tm * tn, 8);
  gemm_bf16_wmma<<<blocks, 256, 0, s>>>(A, lda, aMode, aIdx, aIdxStride, aIdxOff, aRelu,
                                        B, ldb, bias, C, ldc, cColOff, cMode, cSig,
                                        M, N, K, tm, tn);
}

extern "C" void kernel_launch(void* const* d_in, const int* in_sizes, int n_in,
                              void* d_out, int out_size, void* d_ws, size_t ws_size,
                              hipStream_t stream) {
  const float* features   = (const float*)d_in[0];
  const int*   pair_idx   = (const int*)  d_in[1];
  const int*   im_idx     = (const int*)  d_in[2];
  const int*   labels     = (const int*)  d_in[3];
  const float* union_feat = (const float*)d_in[4];
  const float* spatial    = (const float*)d_in[5];
  const float* w_union    = (const float*)d_in[6];
  const float* b_union    = (const float*)d_in[7];
  const float* conv1_w    = (const float*)d_in[8];
  const float* conv1_b    = (const float*)d_in[9];
  const float* bn1_g      = (const float*)d_in[10];
  const float* bn1_b      = (const float*)d_in[11];
  const float* bn1_m      = (const float*)d_in[12];
  const float* bn1_v      = (const float*)d_in[13];
  const float* conv2_w    = (const float*)d_in[14];
  const float* conv2_b    = (const float*)d_in[15];
  const float* bn2_g      = (const float*)d_in[16];
  const float* bn2_b      = (const float*)d_in[17];
  const float* bn2_m      = (const float*)d_in[18];
  const float* bn2_v      = (const float*)d_in[19];
  const float* subj_w     = (const float*)d_in[20];
  const float* subj_b     = (const float*)d_in[21];
  const float* obj_w      = (const float*)d_in[22];
  const float* obj_b      = (const float*)d_in[23];
  const float* vr_w       = (const float*)d_in[24];
  const float* vr_b       = (const float*)d_in[25];
  const float* emb1       = (const float*)d_in[26];
  const float* emb2       = (const float*)d_in[27];
  const float* bil_w      = (const float*)d_in[28];
  const float* bil_b      = (const float*)d_in[29];
  const float* lin_w      = (const float*)d_in[30];
  const float* lin_b      = (const float*)d_in[31];
  const float* proj_w     = (const float*)d_in[32];
  const float* proj_b     = (const float*)d_in[33];
  const float* ac_w       = (const float*)d_in[34];
  const float* ac_b       = (const float*)d_in[35];

  // ---- workspace carve (floats) ----
  float* ws = (float*)d_ws;
  size_t off = 0;
  float* subj   = ws + off; off += 256 * 512;        // [256,512]
  float* objr   = ws + off; off += 256 * 512;        // [256,512]
  float* u_nchw = ws + off; off += 256 * 256 * 49;   // [256,256,49]
  float* c1out  = ws + off; off += 256 * 128 * 196;  // [256,128,14,14]
  float* p1out  = ws + off; off += 256 * 128 * 49;   // [256,128,7,7]
  float* vrin   = ws + off; off += 256 * 256 * 49;   // u + x, [256,12544]
  float* so_cat = ws + off; off += 256 * 1424;       // [256,1424]
  float* zb     = ws + off; off += 256 * 1424;       // [z | bil]
  float* rel    = ws + off; off += 256 * DREL;       // [z2 | vr]
  int* counts = (int*)(ws + off); off += 16;
  int* starts = (int*)(ws + off); off += 16;

  // ---- output carve ----
  float* out_act    = (float*)d_out;                         // 16*157
  float* out_pooled = out_act + 16 * 157;                    // 16*1936
  float* out_mem    = out_pooled + 16 * DREL;                // 24*16*1936
  float* out_mask   = out_mem + L_MAX * B_IMG * DREL;        // 16*24

  // ---- spatial/conv branch ----
  conv1_relu_bn<<<cdiv(256 * 128 * 196, 256), 256, 0, stream>>>(
      spatial, conv1_w, conv1_b, bn1_g, bn1_b, bn1_m, bn1_v, c1out);
  maxpool_3s2<<<cdiv(256 * 128 * 49, 256), 256, 0, stream>>>(c1out, p1out);

  // ---- union 1x1 conv as WMMA GEMM: M=12544 (n,hw), N=256 (oc), K=1024 (c) ----
  run_gemm(stream, union_feat, 1024, /*aMode=*/2, nullptr, 0, 0, 0,
           w_union, 1024, b_union, u_nchw, 0, 0, /*cMode=*/1, 0,
           12544, 256, 1024);

  conv2_relu_bn_add<<<cdiv(256 * 256 * 49, 256), 256, 0, stream>>>(
      p1out, conv2_w, conv2_b, bn2_g, bn2_b, bn2_m, bn2_v, u_nchw, vrin);

  // ---- gathered subj/obj linears: [256,2048]x[2048,512] ----
  run_gemm(stream, features, 2048, /*aMode=*/1, pair_idx, 2, 0, 0,
           subj_w, 2048, subj_b, subj, 512, 0, 0, 0, 256, 512, 2048);
  run_gemm(stream, features, 2048, /*aMode=*/1, pair_idx, 2, 1, 0,
           obj_w, 2048, obj_b, objr, 512, 0, 0, 0, 256, 512, 2048);

  // ---- vr linear: [256,12544]x[12544,512] -> rel[:,1424:1936] ----
  run_gemm(stream, vrin, 12544, 0, nullptr, 0, 0, 0,
           vr_w, 12544, vr_b, rel, DREL, 1424, 0, 0, 256, 512, 12544);

  // ---- s/o concat ----
  build_socat<<<cdiv(256 * 1424, 256), 256, 0, stream>>>(
      subj, objr, emb1, emb2, pair_idx, labels, so_cat);

  // ---- bilinear (dominant op): grid = 16 n-blocks x 712 o ----
  bilinear_wmma<<<dim3(16, 712), 256, 0, stream>>>(so_cat, bil_w, bil_b, zb);

  // ---- lin: [256,1424]x[1424,712] -> zb[:,0:712] ----
  run_gemm(stream, so_cat, 1424, 0, nullptr, 0, 0, 0,
           lin_w, 1424, lin_b, zb, 1424, 0, 0, 0, 256, 712, 1424);

  // ---- proj with relu-on-load of concat(z,bil): -> rel[:,0:1424] ----
  run_gemm(stream, zb, 1424, 0, nullptr, 0, 0, /*aRelu=*/1,
           proj_w, 1424, proj_b, rel, DREL, 0, 0, 0, 256, 1424, 1424);

  // ---- segment pooling / scatter / mask ----
  seg_counts<<<1, 32, 0, stream>>>(im_idx, counts, starts);
  seg_max_pool<<<cdiv(B_IMG * DREL, 256), 256, 0, stream>>>(rel, starts, counts, out_pooled);
  zero_f32<<<cdiv(L_MAX * B_IMG * DREL, 256), 256, 0, stream>>>(out_mem, L_MAX * B_IMG * DREL);
  scatter_memory<<<cdiv(P_PAIRS * DREL, 256), 256, 0, stream>>>(rel, im_idx, starts, out_mem);
  build_mask<<<cdiv(B_IMG * L_MAX, 256), 256, 0, stream>>>(counts, out_mask);

  // ---- act = sigmoid(pooled @ ac_w^T + ac_b): [16,1936]x[1936,157] ----
  run_gemm(stream, out_pooled, DREL, 0, nullptr, 0, 0, 0,
           ac_w, DREL, ac_b, out_act, 157, 0, 0, /*cSig=*/1, 16, 157, 1936);
}
